// WindowAttention_27891517620497
// MI455X (gfx1250) — compile-verified, matrix operand order checked by reading, not run on hardware
//
#include <hip/hip_runtime.h>
#include <hip/hip_bf16.h>

typedef __attribute__((ext_vector_type(16))) _Float16 v16h;
typedef __attribute__((ext_vector_type(8)))  _Float16 v8h;
typedef __attribute__((ext_vector_type(8)))  float    v8f;
typedef __attribute__((ext_vector_type(4)))  float    v4f;

#define N_TOK 343
#define N_PAD 352
#define DIMC  192
#define NHEAD 6
#define HD    32
#define NWIN  64
#define NB    256
#define QT_COUNT 22   /* 352/16 */

union V16U { v16h v; v8h h[2]; };

// ---------------------------------------------------------------------------
// Operand loader for f16 row-major [rows][ld]:
//   A-operand: lane(0..15)=row rowBase+l, k = k0+{0..7,16..23}
//              lane(16..31)=row rowBase+(l-16), k = k0+{8..15,24..31}
// (B-operand uses the same pattern with "row" = output column / key index.)
// All chunks are 16B aligned given ld%8==0 and k0%32==0.
// ---------------------------------------------------------------------------
__device__ __forceinline__ v16h load_f16_tile(const _Float16* base, int rowBase,
                                              int ld, int k0) {
  int l = threadIdx.x;
  int r = l & 15;
  int g = (l >> 4) & 1;
  const _Float16* p = base + (size_t)(rowBase + r) * ld + k0 + g * 8;
  V16U u;
  u.h[0] = *(const v8h*)(p);
  u.h[1] = *(const v8h*)(p + 16);
  return u.v;
}

// Same lane mapping, but source is f32 (converted to f16); rows >= nvalid -> 0
__device__ __forceinline__ v16h load_f32_tile_cvt(const float* base, int rowBase,
                                                  int nvalid, int ld, int k0) {
  int l = threadIdx.x;
  int r = l & 15;
  int g = (l >> 4) & 1;
  int row = rowBase + r;
  V16U u;
  if (row < nvalid) {
    const float* p = base + (size_t)row * ld + k0 + g * 8;
    v4f a0 = *(const v4f*)(p);
    v4f a1 = *(const v4f*)(p + 4);
    v4f b0 = *(const v4f*)(p + 16);
    v4f b1 = *(const v4f*)(p + 20);
#pragma unroll
    for (int i = 0; i < 4; ++i) {
      u.h[0][i]     = (_Float16)a0[i];
      u.h[0][4 + i] = (_Float16)a1[i];
      u.h[1][i]     = (_Float16)b0[i];
      u.h[1][4 + i] = (_Float16)b1[i];
    }
  } else {
#pragma unroll
    for (int i = 0; i < 8; ++i) { u.h[0][i] = (_Float16)0.f; u.h[1][i] = (_Float16)0.f; }
  }
  return u.v;
}

__device__ __forceinline__ v8f wmma_f16(v16h a, v16h b, v8f c) {
  return __builtin_amdgcn_wmma_f32_16x16x32_f16(false, a, false, b, (short)0, c,
                                                false, false);
}

// ---------------------------------------------------------------------------
// Stage 0a: f32 -> f16 weight conversion
// ---------------------------------------------------------------------------
__global__ void cvt_f32_to_f16(const float* __restrict__ src,
                               _Float16* __restrict__ dst, int n) {
  int i = blockIdx.x * 256 + threadIdx.x;
  if (i < n) dst[i] = (_Float16)src[i];
}

// ---------------------------------------------------------------------------
// Stage 0b: relative-position bias gather: biasM[h][q][k] = table[rel_idx[q][k]][h]
// ---------------------------------------------------------------------------
__global__ void bias_gather(const float* __restrict__ table,
                            const int* __restrict__ rel_idx,
                            float* __restrict__ biasM) {
  int i = blockIdx.x * 256 + threadIdx.x;
  if (i < N_TOK * N_TOK) {
    int idx = rel_idx[i];
#pragma unroll
    for (int h = 0; h < NHEAD; ++h)
      biasM[(size_t)h * N_TOK * N_TOK + i] = table[idx * NHEAD + h];
  }
}

// ---------------------------------------------------------------------------
// Stage 1: QKV projection GEMM (one wave per 16x16 output tile).
//   blockIdx.x = n-tile (22), blockIdx.y = c-tile (12 Q + 24 KV), blockIdx.z = b
//   Q  -> Qh[b][h][n][d]  (f16, pre-scaled by hd^-0.5)
//   K  -> Kh[b][h][n][d]  (f16)
//   V  -> Vt[b][h][d][n]  (f16, transposed; contiguous stores)
// ---------------------------------------------------------------------------
__global__ __launch_bounds__(32) void qkv_gemm(
    const float* __restrict__ x1, const float* __restrict__ x2,
    const _Float16* __restrict__ qwh, const _Float16* __restrict__ kvwh,
    const float* __restrict__ qb, const float* __restrict__ kvb,
    _Float16* __restrict__ Qh, _Float16* __restrict__ Kh,
    _Float16* __restrict__ Vt) {
  int nt = blockIdx.x;
  int ct = blockIdx.y;
  int b  = blockIdx.z;
  bool isQ = (ct < 12);
  int cbase = (isQ ? ct : (ct - 12)) * 16;
  const float*    X = isQ ? x1 : x2;
  const _Float16* W = isQ ? qwh : kvwh;
  int nbase = nt * 16;

  v8f acc = {};
#pragma unroll
  for (int k0 = 0; k0 < DIMC; k0 += 32) {
    v16h a = load_f32_tile_cvt(X + (size_t)b * N_TOK * DIMC, nbase, N_TOK, DIMC, k0);
    v16h w = load_f16_tile(W, cbase, DIMC, k0);
    acc = wmma_f16(a, w, acc);
  }

  int l = threadIdx.x;
  int g = (l >> 4) & 1;           // register group: rows nbase+8g .. nbase+8g+7
  int c = cbase + (l & 15);       // this lane's output column

  if (isQ) {
    const float scale = 0.17677669529663687f;  // 32^-0.5
    float bias = qb[c];
    int h = c >> 5, d = c & 31;
    _Float16* base = Qh + ((size_t)b * NHEAD + h) * N_PAD * HD;
#pragma unroll
    for (int r = 0; r < 8; ++r) {
      int n = nbase + 8 * g + r;
      base[(size_t)n * HD + d] = (_Float16)((acc[r] + bias) * scale);
    }
  } else {
    float bias = kvb[c];
    bool isK = (c < DIMC);
    int c2 = isK ? c : c - DIMC;
    int h = c2 >> 5, d = c2 & 31;
    if (isK) {
      _Float16* base = Kh + ((size_t)b * NHEAD + h) * N_PAD * HD;
#pragma unroll
      for (int r = 0; r < 8; ++r) {
        int n = nbase + 8 * g + r;
        base[(size_t)n * HD + d] = (_Float16)(acc[r] + bias);
      }
    } else {
      _Float16* dst = Vt + (((size_t)b * NHEAD + h) * HD + d) * N_PAD + nbase + 8 * g;
      v8h o;
#pragma unroll
      for (int r = 0; r < 8; ++r) o[r] = (_Float16)(acc[r] + bias);
      *(v8h*)dst = o;   // 16B contiguous
    }
  }
}

// ---------------------------------------------------------------------------
// Stage 2: flash attention, one wave per (b, h, q-tile).
//   Computes S^T = K x Q^T so P^T (in D-register layout) is directly the
//   B operand of O^T = V^T x P^T (32 keys per iteration => K=32 exactly).
// ---------------------------------------------------------------------------
__global__ __launch_bounds__(32) void flash_attn(
    const _Float16* __restrict__ Qh, const _Float16* __restrict__ Kh,
    const _Float16* __restrict__ Vt, const float* __restrict__ biasM,
    const float* __restrict__ mask, _Float16* __restrict__ att) {
  int qt = blockIdx.x;            // 0..21
  int h  = blockIdx.y;            // 0..5
  int b  = blockIdx.z;            // 0..255
  int w  = b & (NWIN - 1);        // window index for mask

  const _Float16* Qb = Qh + ((size_t)b * NHEAD + h) * N_PAD * HD;
  const _Float16* Kb = Kh + ((size_t)b * NHEAD + h) * N_PAD * HD;
  const _Float16* Vb = Vt + ((size_t)b * NHEAD + h) * HD * N_PAD;
  const float* Bm = biasM + (size_t)h * N_TOK * N_TOK;
  const float* Mm = mask + (size_t)w * N_TOK * N_TOK;

  int l  = threadIdx.x;
  int q  = l & 15;
  int g  = (l >> 4) & 1;
  int qg = qt * 16 + q;           // this lane's global query index

  v16h Qop = load_f16_tile(Qb, qt * 16, HD, 0);   // B operand, loaded once

  v8f oLo = {};                   // O^T rows d=0..15 (this lane: d=8g+r)
  v8f oHi = {};                   // O^T rows d=16..31
  float mrow = -3.0e38f;
  float lsum = 0.f;

  for (int k0 = 0; k0 < N_PAD; k0 += 32) {
    v16h Ka = load_f16_tile(Kb, k0, HD, 0);
    v16h Kc = load_f16_tile(Kb, k0 + 16, HD, 0);
    v8f z0 = {};
    v8f z1 = {};
    v8f s0 = wmma_f16(Ka, Qop, z0);   // keys k0   .. k0+15 (x query)
    v8f s1 = wmma_f16(Kc, Qop, z1);   // keys k0+16.. k0+31

    int kb0 = k0 + 8 * g;           // s0 reg r -> key kb0+r
    int kb1 = k0 + 16 + 8 * g;      // s1 reg r -> key kb1+r
    if (qg < N_TOK) {
      const float* bp = Bm + (size_t)qg * N_TOK;
      const float* mp = Mm + (size_t)qg * N_TOK;
#pragma unroll
      for (int r = 0; r < 8; ++r) {
        int k1 = kb0 + r;
        s0[r] = (k1 < N_TOK) ? (s0[r] + bp[k1] + mp[k1]) : -1.0e30f;
        int k2 = kb1 + r;
        s1[r] = (k2 < N_TOK) ? (s1[r] + bp[k2] + mp[k2]) : -1.0e30f;
      }
    } else {
#pragma unroll
      for (int r = 0; r < 8; ++r) {
        if (kb0 + r >= N_TOK) s0[r] = -1.0e30f;
        if (kb1 + r >= N_TOK) s1[r] = -1.0e30f;
      }
    }

    // --- online softmax (row = query; pair (lane, lane^16) shares a row) ---
    float tm = s0[0];
#pragma unroll
    for (int r = 0; r < 8; ++r) { tm = fmaxf(tm, s0[r]); tm = fmaxf(tm, s1[r]); }
    tm = fmaxf(tm, __shfl_xor(tm, 16, 32));
    float mnew  = fmaxf(mrow, tm);
    float alpha = __expf(mrow - mnew);
    mrow = mnew;

    V16U p;                       // P^T in B-operand layout for K=32
    float rs = 0.f;
#pragma unroll
    for (int r = 0; r < 8; ++r) {
      float e0 = __expf(s0[r] - mnew);
      float e1 = __expf(s1[r] - mnew);
      rs += e0 + e1;
      p.h[0][r] = (_Float16)e0;   // keys kb0+r
      p.h[1][r] = (_Float16)e1;   // keys kb1+r
    }
    rs += __shfl_xor(rs, 16, 32);
    lsum = lsum * alpha + rs;

#pragma unroll
    for (int i = 0; i < 8; ++i) { oLo[i] *= alpha; oHi[i] *= alpha; }

    v16h Va0 = load_f16_tile(Vb, 0,  N_PAD, k0);  // V^T rows d=0..15
    v16h Va1 = load_f16_tile(Vb, 16, N_PAD, k0);  // V^T rows d=16..31
    oLo = wmma_f16(Va0, p.v, oLo);
    oHi = wmma_f16(Va1, p.v, oHi);
  }

  if (qg < N_TOK) {
    float inv = 1.0f / lsum;
    _Float16* dst = att + ((size_t)b * N_PAD + qg) * DIMC + h * HD;
    v8h a0, a1;
#pragma unroll
    for (int r = 0; r < 8; ++r) {
      a0[r] = (_Float16)(oLo[r] * inv);   // d = 8g + r
      a1[r] = (_Float16)(oHi[r] * inv);   // d = 16 + 8g + r
    }
    *(v8h*)(dst + 8 * g)      = a0;       // 16B contiguous
    *(v8h*)(dst + 16 + 8 * g) = a1;
  }
}

// ---------------------------------------------------------------------------
// Stage 3: output projection GEMM. out[b][n][c] = att[b][n][:] . proj_w[c][:] + pb[c]
// ---------------------------------------------------------------------------
__global__ __launch_bounds__(32) void proj_gemm(
    const _Float16* __restrict__ att, const _Float16* __restrict__ pwh,
    const float* __restrict__ pb, float* __restrict__ out) {
  int nt = blockIdx.x;   // 22
  int ct = blockIdx.y;   // 12
  int b  = blockIdx.z;   // 256

  v8f acc = {};
#pragma unroll
  for (int k0 = 0; k0 < DIMC; k0 += 32) {
    v16h a = load_f16_tile(att + (size_t)b * N_PAD * DIMC, nt * 16, DIMC, k0);
    v16h wv = load_f16_tile(pwh, ct * 16, DIMC, k0);
    acc = wmma_f16(a, wv, acc);
  }

  int l = threadIdx.x;
  int g = (l >> 4) & 1;
  int c = ct * 16 + (l & 15);
  float bias = pb[c];
  float* dst = out + (size_t)b * N_TOK * DIMC + c;
#pragma unroll
  for (int r = 0; r < 8; ++r) {
    int n = nt * 16 + 8 * g + r;
    if (n < N_TOK) dst[(size_t)n * DIMC] = acc[r] + bias;
  }
}

// ---------------------------------------------------------------------------
// Host launcher
// ---------------------------------------------------------------------------
extern "C" void kernel_launch(void* const* d_in, const int* in_sizes, int n_in,
                              void* d_out, int out_size, void* d_ws, size_t ws_size,
                              hipStream_t stream) {
  const float* x1     = (const float*)d_in[0];
  const float* x2     = (const float*)d_in[1];
  const float* mask   = (const float*)d_in[2];
  const float* table  = (const float*)d_in[3];
  const float* q_w    = (const float*)d_in[4];
  const float* q_b    = (const float*)d_in[5];
  const float* kv_w   = (const float*)d_in[6];
  const float* kv_b   = (const float*)d_in[7];
  const float* proj_w = (const float*)d_in[8];
  const float* proj_b = (const float*)d_in[9];
  const int*   relidx = (const int*)d_in[10];
  float* out = (float*)d_out;

  char* ws = (char*)d_ws;
  // workspace layout (256B aligned)
  const size_t o_qwh  = 0;                                 //  73,728 B
  const size_t o_kvwh = o_qwh  + 73728;                    // 147,456 B
  const size_t o_pwh  = o_kvwh + 147456;                   //  73,728 B
  const size_t o_bias = o_pwh  + 73728;                    // 2,823,680 B (padded)
  const size_t o_Qh   = o_bias + 2823680;                  // 34,603,008 B
  const size_t o_Kh   = o_Qh   + 34603008;
  const size_t o_Vt   = o_Kh   + 34603008;
  const size_t o_att  = o_Vt   + 34603008;

  _Float16* qwh  = (_Float16*)(ws + o_qwh);
  _Float16* kvwh = (_Float16*)(ws + o_kvwh);
  _Float16* pwh  = (_Float16*)(ws + o_pwh);
  float*    bsm  = (float*)   (ws + o_bias);
  _Float16* Qh   = (_Float16*)(ws + o_Qh);
  _Float16* Kh   = (_Float16*)(ws + o_Kh);
  _Float16* Vt   = (_Float16*)(ws + o_Vt);
  _Float16* att  = (_Float16*)(ws + o_att);

  // Stage 0: weight conversion + bias gather
  cvt_f32_to_f16<<<(192 * 192 + 255) / 256, 256, 0, stream>>>(q_w, qwh, 192 * 192);
  cvt_f32_to_f16<<<(384 * 192 + 255) / 256, 256, 0, stream>>>(kv_w, kvwh, 384 * 192);
  cvt_f32_to_f16<<<(192 * 192 + 255) / 256, 256, 0, stream>>>(proj_w, pwh, 192 * 192);
  bias_gather<<<(N_TOK * N_TOK + 255) / 256, 256, 0, stream>>>(table, relidx, bsm);

  // Stage 1: QKV projections (f16 WMMA, f32 accumulate)
  qkv_gemm<<<dim3(QT_COUNT, 12 + 24, NB), 32, 0, stream>>>(
      x1, x2, qwh, kvwh, q_b, kv_b, Qh, Kh, Vt);

  // Stage 2: fused flash attention (QK^T + bias + mask + softmax + PV)
  flash_attn<<<dim3(QT_COUNT, NHEAD, NB), 32, 0, stream>>>(
      Qh, Kh, Vt, bsm, mask, att);

  // Stage 3: output projection
  proj_gemm<<<dim3(QT_COUNT, 12, NB), 32, 0, stream>>>(att, pwh, proj_b, out);
}
